// Int8Linear_25237227831506
// MI455X (gfx1250) — matrix operand order repaired and checked by reading, not compiled, not run on hardware
//
#include <hip/hip_runtime.h>

typedef __attribute__((ext_vector_type(16))) _Float16 v16h;
typedef __attribute__((ext_vector_type(8)))  _Float16 v8h;
typedef __attribute__((ext_vector_type(4)))  _Float16 v4h;
typedef __attribute__((ext_vector_type(8)))  float    v8f;
typedef __attribute__((ext_vector_type(4)))  float    v4f;

#define DIN   4096
#define DOUT  16384
#define MTOT  8192          // B * S
#define BM    128
#define BN    128
#define BK    64
#define NK    (DIN / BK)    // 64 K-iterations

// XOR-swizzled LDS layout: row stride = 64 halves (128 B = 32 dwords), with the
// 16-byte chunk index XOR'd by row bits so that the 16 rows read together by a
// WMMA fragment load (ds_load_b128) hit 16 distinct 4-dword bank groups.
// hc = half-index within the logical row.
__device__ __forceinline__ int swz(int row, int hc) {
  int chunk = (hc >> 3) ^ ((row >> 1) & 7);
  return row * BK + (chunk << 3) + (hc & 7);
}

__global__ __launch_bounds__(256) void int8linear_wmma(
    const float* __restrict__ x,          // [MTOT][DIN] fp32
    const signed char* __restrict__ w,    // [DOUT][DIN] int8  (== B^T, i.e. [N][K])
    const float* __restrict__ scale,      // [DOUT]
    const float* __restrict__ bias,       // [DOUT]
    float* __restrict__ out)              // [MTOT][DOUT] fp32
{
  // 64 KB total: double-buffered 128x64 f16 tiles for A and B.
  __shared__ _Float16 ldsA[2][BM * BK];
  __shared__ _Float16 ldsB[2][BN * BK];

  const int tid  = threadIdx.x;
  const int lane = tid & 31;
  const int wave = tid >> 5;
  const int lr   = lane & 15;   // row/col within fragment
  const int lh   = lane >> 4;   // half-wave selector

  const int gm = blockIdx.y * BM;
  const int gn = blockIdx.x * BN;

  // A-tile fill mapping: 128x64 fp32 -> 8 float4 per thread
  const int ar = tid >> 4;      // 0..15 (row group)
  const int ac = tid & 15;      // 0..15 (float4 column)
  // B-tile fill mapping: 128x64 int8 -> 2 x 16B per thread
  const int br = tid >> 2;      // 0..63 (row group)
  const int bc = tid & 3;       // 0..3  (16-byte column)

  // 4x2 wave grid; each wave owns a 32x64 sub-tile = 2x4 WMMA accumulators
  const int wm = (wave >> 1) * 32;
  const int wn = (wave & 1) * 64;

  v8f acc[2][4];
#pragma unroll
  for (int i = 0; i < 2; ++i)
#pragma unroll
    for (int j = 0; j < 4; ++j) acc[i][j] = (v8f)0.0f;

  v4f areg[8];
  int4 breg[2];

  auto gload = [&](int kt) {
    const float* ap = x + (size_t)gm * DIN + (size_t)kt * BK + ac * 4;
#pragma unroll
    for (int i = 0; i < 8; ++i)
      areg[i] = *(const v4f*)(ap + (size_t)(i * 16 + ar) * DIN);
    const signed char* bp = w + (size_t)gn * DIN + (size_t)kt * BK + bc * 16;
#pragma unroll
    for (int i = 0; i < 2; ++i)
      breg[i] = *(const int4*)(bp + (size_t)(i * 64 + br) * DIN);
  };

  auto sts = [&](int buf) {
    // A: fp32 -> f16, 4 halves (8 B) per store
#pragma unroll
    for (int i = 0; i < 8; ++i) {
      const int row = i * 16 + ar;
      v4h h;
      h[0] = (_Float16)areg[i][0];
      h[1] = (_Float16)areg[i][1];
      h[2] = (_Float16)areg[i][2];
      h[3] = (_Float16)areg[i][3];
      *(v4h*)&ldsA[buf][swz(row, ac * 4)] = h;
    }
    // B: int8 -> f16 (exact), 16 halves per 16 source bytes
#pragma unroll
    for (int i = 0; i < 2; ++i) {
      const int row = i * 64 + br;
      union { int4 v; signed char c[16]; } u;
      u.v = breg[i];
      v8h h0, h1;
#pragma unroll
      for (int j = 0; j < 8; ++j) {
        h0[j] = (_Float16)u.c[j];
        h1[j] = (_Float16)u.c[8 + j];
      }
      *(v8h*)&ldsB[buf][swz(row, bc * 16)]     = h0;
      *(v8h*)&ldsB[buf][swz(row, bc * 16 + 8)] = h1;
    }
  };

  auto compute = [&](int buf) {
#pragma unroll
    for (int kk = 0; kk < 2; ++kk) {           // two k=32 WMMA slabs per BK=64 tile
      v16h afrag[2];
#pragma unroll
      for (int mt = 0; mt < 2; ++mt) {
        // ISA 16-bit A 16x32 layout: lane<16 holds K {0..7,16..23}, lane>=16 holds
        // K {8..15,24..31} -> two 16B chunks per lane.
        const int row = wm + mt * 16 + lr;
        v8h lo = *(const v8h*)&ldsA[buf][swz(row, kk * 32 + lh * 8)];
        v8h hi = *(const v8h*)&ldsA[buf][swz(row, kk * 32 + 16 + lh * 8)];
        afrag[mt] = __builtin_shufflevector(lo, hi, 0, 1, 2, 3, 4, 5, 6, 7,
                                            8, 9, 10, 11, 12, 13, 14, 15);
      }
#pragma unroll
      for (int nt = 0; nt < 4; ++nt) {
        // B 32x16 layout: lane<16 -> K {0..7,16..23}, lane>=16 -> K {8..15,24..31}, N = lane%16.
        // ldsB rows are already [N][K] (W is B-transposed in memory).
        const int row = wn + nt * 16 + lr;
        v8h lo = *(const v8h*)&ldsB[buf][swz(row, kk * 32 + lh * 8)];
        v8h hi = *(const v8h*)&ldsB[buf][swz(row, kk * 32 + 16 + lh * 8)];
        v16h bfrag = __builtin_shufflevector(lo, hi, 0, 1, 2, 3, 4, 5, 6, 7,
                                             8, 9, 10, 11, 12, 13, 14, 15);
#pragma unroll
        for (int mt = 0; mt < 2; ++mt) {
          acc[mt][nt] = __builtin_amdgcn_wmma_f32_16x16x32_f16(
              false, afrag[mt], false, bfrag, (short)0, acc[mt][nt],
              false, false);
        }
      }
    }
  };

  gload(0);
#pragma unroll 1
  for (int kt = 0; kt < NK; ++kt) {
    const int buf = kt & 1;
    sts(buf);
    __syncthreads();                 // single barrier per iter (double-buffered)
    if (kt + 1 < NK) gload(kt + 1);  // prefetch next tile behind the WMMAs
    compute(buf);
  }

  // Epilogue: dequant scale + bias in f32, per C/D layout:
  // VGPR i, lanes 0-15 -> row i, N=lane; lanes 16-31 -> row i+8, N=lane-16.
#pragma unroll
  for (int nt = 0; nt < 4; ++nt) {
    const int col = gn + wn + nt * 16 + lr;
    const float s = scale[col];
    const float b = bias[col];
#pragma unroll
    for (int mt = 0; mt < 2; ++mt) {
      const int rowbase = gm + wm + mt * 16 + lh * 8;
#pragma unroll
      for (int i = 0; i < 8; ++i) {
        out[(size_t)(rowbase + i) * DOUT + col] = acc[mt][nt][i] * s + b;
      }
    }
  }
}

extern "C" void kernel_launch(void* const* d_in, const int* in_sizes, int n_in,
                              void* d_out, int out_size, void* d_ws, size_t ws_size,
                              hipStream_t stream) {
  const float*       x  = (const float*)d_in[0];
  const signed char* w  = (const signed char*)d_in[1];
  const float*       sc = (const float*)d_in[2];
  const float*       bi = (const float*)d_in[3];
  float*             y  = (float*)d_out;

  dim3 grid(DOUT / BN, MTOT / BM);   // (128, 64)
  int8linear_wmma<<<grid, 256, 0, stream>>>(x, w, sc, bi, y);
}
